// Discriminator_6622839571270
// MI455X (gfx1250) — compile-verified, matrix operand order checked by reading, not run on hardware
//
#include <hip/hip_runtime.h>
#include <hip/hip_bf16.h>
#include <math.h>

// ---------------------------------------------------------------------------
// Curb-GAN Discriminator forward for MI455X (gfx1250, wave32, WMMA).
// All dense GEMMs go through v_wmma_f32_16x16x32_f16 (f16 in, f32 accum).
// Inner loops are branch-free: vectorized b128 A loads, clamped-address B
// loads (no EXEC manipulation), guarded work only in the K%32 tail chunk.
// ---------------------------------------------------------------------------

typedef __attribute__((ext_vector_type(16))) _Float16 v16h;
typedef __attribute__((ext_vector_type(8)))  float    v8f;

#define BT    3072      // B*T = 256*12
#define NN    100
#define ROWS  307200    // BT*NN
#define TT    12

__device__ inline float warp_sum(float v) {
#pragma unroll
  for (int m = 16; m >= 1; m >>= 1) v += __shfl_xor(v, m, 32);
  return v;
}

// Fragment index -> K offset for 16-bit A (16x32) / mirrored B, wave32.
// element e (0..15), half = lane>>4:  k = (e>>3)*16 + half*8 + (e&7)
__device__ inline int frag_k(int e, int half) {
  return ((e >> 3) << 4) + (half << 3) + (e & 7);
}

__device__ inline float bn_leaky(float v, float s1, float s0, int do_bn) {
  v = fmaf(v, s1, s0);
  if (do_bn) v = v >= 0.f ? v : 0.2f * v;   // leaky_relu(0.2)
  return v;
}

// ---------------------------------------------------------------------------
// pack h0 = concat(x, c) -> X[ROWS][4]
// ---------------------------------------------------------------------------
__global__ void pack_h0_kernel(const float* __restrict__ x,
                               const float* __restrict__ c,
                               float* __restrict__ X) {
  int r = blockIdx.x * 256 + threadIdx.x;
  if (r >= ROWS) return;
  X[r * 4 + 0] = x[r];
  X[r * 4 + 1] = c[r * 3 + 0];
  X[r * 4 + 2] = c[r * 3 + 1];
  X[r * 4 + 3] = c[r * 3 + 2];
}

// ---------------------------------------------------------------------------
// HW = normleaky(raw) @ W   : (ROWS x K) @ (K x N) -> f16 (ROWS x N)
// BN of the PREVIOUS layer applied on A-load (per (t,n) stats, per-n gamma/beta)
// grid = (Mtiles/4, Ntiles), block = 128 (4 waves, wave per 16x16 tile)
// ---------------------------------------------------------------------------
__global__ void gemm_hw_kernel(const float* __restrict__ Araw,
                               const float* __restrict__ W,
                               _Float16* __restrict__ HW,
                               const float* __restrict__ mu,
                               const float* __restrict__ rstd,
                               const float* __restrict__ gamma,
                               const float* __restrict__ beta,
                               int M, int N, int K, int do_bn) {
  int wave = threadIdx.x >> 5;
  int lane = threadIdx.x & 31;
  int mt = blockIdx.x * 4 + wave;
  int nt = blockIdx.y;
  int half = lane >> 4, l16 = lane & 15;
  int row = mt * 16 + l16;
  int col = nt * 16 + l16;
  bool rok = row < M, cok = col < N;
  int rowc = rok ? row : M - 1;
  int colc = cok ? col : N - 1;

  // BN folded to affine: v' = v*s1 + s0 (then leaky if do_bn)
  float s1 = 1.f, s0 = 0.f;
  if (do_bn) {
    int n = rowc % NN;
    int t = (rowc / NN) % TT;
    float g = rstd[t * NN + n] * gamma[n];
    s1 = g;
    s0 = beta[n] - mu[t * NN + n] * g;
  }

  const float* Ar = Araw + (long)rowc * K;

  v8f acc;
#pragma unroll
  for (int i = 0; i < 8; ++i) acc[i] = 0.f;

  int kmain = K & ~31;
  for (int k0 = 0; k0 < kmain; k0 += 32) {
    const float4* p0 = (const float4*)(Ar + k0 + half * 8);
    const float4* p1 = (const float4*)(Ar + k0 + 16 + half * 8);
    float4 a0 = p0[0], a1 = p0[1], a2 = p1[0], a3 = p1[1];
    v16h a, b;
#pragma unroll
    for (int e = 0; e < 16; ++e)
      b[e] = (_Float16)W[(long)(k0 + frag_k(e, half)) * N + colc];
    a[0]  = (_Float16)bn_leaky(a0.x, s1, s0, do_bn);
    a[1]  = (_Float16)bn_leaky(a0.y, s1, s0, do_bn);
    a[2]  = (_Float16)bn_leaky(a0.z, s1, s0, do_bn);
    a[3]  = (_Float16)bn_leaky(a0.w, s1, s0, do_bn);
    a[4]  = (_Float16)bn_leaky(a1.x, s1, s0, do_bn);
    a[5]  = (_Float16)bn_leaky(a1.y, s1, s0, do_bn);
    a[6]  = (_Float16)bn_leaky(a1.z, s1, s0, do_bn);
    a[7]  = (_Float16)bn_leaky(a1.w, s1, s0, do_bn);
    a[8]  = (_Float16)bn_leaky(a2.x, s1, s0, do_bn);
    a[9]  = (_Float16)bn_leaky(a2.y, s1, s0, do_bn);
    a[10] = (_Float16)bn_leaky(a2.z, s1, s0, do_bn);
    a[11] = (_Float16)bn_leaky(a2.w, s1, s0, do_bn);
    a[12] = (_Float16)bn_leaky(a3.x, s1, s0, do_bn);
    a[13] = (_Float16)bn_leaky(a3.y, s1, s0, do_bn);
    a[14] = (_Float16)bn_leaky(a3.z, s1, s0, do_bn);
    a[15] = (_Float16)bn_leaky(a3.w, s1, s0, do_bn);
    acc = __builtin_amdgcn_wmma_f32_16x16x32_f16(false, a, false, b,
                                                 (short)0, acc, false, false);
  }
  if (kmain < K) {            // tail chunk: clamped loads + value selects
    v16h a, b;
#pragma unroll
    for (int e = 0; e < 16; ++e) {
      int k = kmain + frag_k(e, half);
      int kc = k < K ? k : K - 1;
      float av = bn_leaky(Ar[kc], s1, s0, do_bn);
      a[e] = (_Float16)(k < K ? av : 0.f);
      b[e] = (_Float16)W[(long)kc * N + colc];
    }
    acc = __builtin_amdgcn_wmma_f32_16x16x32_f16(false, a, false, b,
                                                 (short)0, acc, false, false);
  }
  if (cok) {
#pragma unroll
    for (int r = 0; r < 8; ++r) {
      int m = mt * 16 + half * 8 + r;
      if (m < M) HW[(long)m * N + col] = (_Float16)acc[r];
    }
  }
}

// ---------------------------------------------------------------------------
// out[bt] = adj[bt] @ HW[bt] + bias : batched 100x100 @ 100xN
// grid = (BT, 7, ceil(Ntiles/4)), block = 128
// ---------------------------------------------------------------------------
__global__ void gemm_adj_kernel(const float* __restrict__ adj,
                                const _Float16* __restrict__ HW,
                                const float* __restrict__ bias,
                                float* __restrict__ out, int N) {
  int bt = blockIdx.x;
  int mt = blockIdx.y;
  int wave = threadIdx.x >> 5;
  int ntiles = (N + 15) >> 4;
  int nt = blockIdx.z * 4 + wave;
  if (nt >= ntiles) nt = 0;   // keep EXEC full for WMMA; duplicate work is benign

  const float*    A = adj + (long)bt * NN * NN;
  const _Float16* B = HW  + (long)bt * NN * N;

  int lane = threadIdx.x & 31;
  int half = lane >> 4, l16 = lane & 15;
  int row = mt * 16 + l16;
  int col = nt * 16 + l16;
  bool cok = col < N;
  int rowc = row < NN ? row : NN - 1;
  int colc = cok ? col : N - 1;
  const float* Ar = A + rowc * NN;

  v8f acc;
#pragma unroll
  for (int i = 0; i < 8; ++i) acc[i] = 0.f;

  // K = 100: three full chunks + tail of 4
  for (int k0 = 0; k0 < 96; k0 += 32) {
    const float4* p0 = (const float4*)(Ar + k0 + half * 8);
    const float4* p1 = (const float4*)(Ar + k0 + 16 + half * 8);
    float4 a0 = p0[0], a1 = p0[1], a2 = p1[0], a3 = p1[1];
    v16h a, b;
#pragma unroll
    for (int e = 0; e < 16; ++e)
      b[e] = B[(long)(k0 + frag_k(e, half)) * N + colc];
    a[0] = (_Float16)a0.x; a[1] = (_Float16)a0.y; a[2] = (_Float16)a0.z; a[3] = (_Float16)a0.w;
    a[4] = (_Float16)a1.x; a[5] = (_Float16)a1.y; a[6] = (_Float16)a1.z; a[7] = (_Float16)a1.w;
    a[8] = (_Float16)a2.x; a[9] = (_Float16)a2.y; a[10] = (_Float16)a2.z; a[11] = (_Float16)a2.w;
    a[12] = (_Float16)a3.x; a[13] = (_Float16)a3.y; a[14] = (_Float16)a3.z; a[15] = (_Float16)a3.w;
    acc = __builtin_amdgcn_wmma_f32_16x16x32_f16(false, a, false, b,
                                                 (short)0, acc, false, false);
  }
  {  // tail: k = 96..99
    v16h a, b;
#pragma unroll
    for (int e = 0; e < 16; ++e) {
      int k = 96 + frag_k(e, half);
      int kc = k < NN ? k : NN - 1;
      float av = Ar[kc];
      a[e] = (_Float16)(k < NN ? av : 0.f);
      b[e] = B[(long)kc * N + colc];
    }
    acc = __builtin_amdgcn_wmma_f32_16x16x32_f16(false, a, false, b,
                                                 (short)0, acc, false, false);
  }
  if (cok) {
    float bia = bias[col];
#pragma unroll
    for (int r = 0; r < 8; ++r) {
      int m = mt * 16 + half * 8 + r;
      if (m < NN) out[((long)bt * NN + m) * N + col] = acc[r] + bia;
    }
  }
}

// ---------------------------------------------------------------------------
// BatchNorm stats: per (t,n) over (batch, feature). grid = 1200, block = 256
// ---------------------------------------------------------------------------
__global__ void bn_stats_kernel(const float* __restrict__ raw,
                                float* __restrict__ mu,
                                float* __restrict__ rstd, int fout) {
  int tn = blockIdx.x;          // t*100 + n
  int t = tn / NN, n = tn % NN;
  int cnt = 256 * fout;
  float s = 0.f, ss = 0.f;
  for (int i = threadIdx.x; i < cnt; i += blockDim.x) {
    int b = i / fout, f = i % fout;
    float v = raw[((long)(b * TT + t) * NN + n) * fout + f];
    s += v; ss += v * v;
  }
  __shared__ float sh0[8], sh1[8];
  s = warp_sum(s); ss = warp_sum(ss);
  int w = threadIdx.x >> 5, ln = threadIdx.x & 31;
  if (ln == 0) { sh0[w] = s; sh1[w] = ss; }
  __syncthreads();
  if (threadIdx.x == 0) {
    float S = 0.f, SS = 0.f;
    for (int i = 0; i < 8; ++i) { S += sh0[i]; SS += sh1[i]; }
    float m = S / cnt;
    float var = SS / cnt - m * m;
    mu[tn] = m;
    rstd[tn] = rsqrtf(var + 1e-5f);
  }
}

// ---------------------------------------------------------------------------
// finalize DyConv layer 4: hseq = leaky(BN(raw)), raw has fout=1
// ---------------------------------------------------------------------------
__global__ void finalize_dy_kernel(const float* __restrict__ raw,
                                   const float* __restrict__ mu,
                                   const float* __restrict__ rstd,
                                   const float* __restrict__ g,
                                   const float* __restrict__ bb,
                                   float* __restrict__ hseq) {
  int r = blockIdx.x * 256 + threadIdx.x;
  if (r >= ROWS) return;
  int n = r % NN;
  int t = (r / NN) % TT;
  float v = raw[r];
  v = (v - mu[t * NN + n]) * rstd[t * NN + n] * g[n] + bb[n];
  hseq[r] = v >= 0.f ? v : 0.2f * v;
}

// ---------------------------------------------------------------------------
// Row LayerNorm (Curb-GAN Norm): a*(x-mu)/(std_ddof1 + 1e-6) + b, rows of 100
// grid = 768, block = 128 (wave per row)
// ---------------------------------------------------------------------------
__global__ void ln_rows_kernel(const float* __restrict__ h,
                               const float* __restrict__ a,
                               const float* __restrict__ b,
                               float* __restrict__ x2) {
  int row = blockIdx.x * 4 + (threadIdx.x >> 5);
  int lane = threadIdx.x & 31;
  const float* hr = h + (long)row * NN;
  float v[4];
  float s = 0.f;
#pragma unroll
  for (int i = 0; i < 4; ++i) {
    int n = lane + 32 * i;
    v[i] = (n < NN) ? hr[n] : 0.f;
    s += v[i];
  }
  s = warp_sum(s);
  float m = s * (1.f / NN);
  float ss = 0.f;
#pragma unroll
  for (int i = 0; i < 4; ++i) {
    int n = lane + 32 * i;
    if (n < NN) { float d = v[i] - m; ss += d * d; }
  }
  ss = warp_sum(ss);
  float sd = sqrtf(ss / (NN - 1));  // unbiased (ddof=1)
  float inv = 1.f / (sd + 1e-6f);
#pragma unroll
  for (int i = 0; i < 4; ++i) {
    int n = lane + 32 * i;
    if (n < NN) x2[(long)row * NN + n] = a[n] * (v[i] - m) * inv + b[n];
  }
}

// ---------------------------------------------------------------------------
// Generic GEMM: C = [resid +] act(A @ W + bias), A:(M,K) W:(K,N) all f32
// grid = (M/64, Ntiles), block = 128.  M, K%32 handled; M must be mult of 16.
// ---------------------------------------------------------------------------
__global__ void gemm_rm_kernel(const float* __restrict__ A,
                               const float* __restrict__ W,
                               const float* __restrict__ bias,
                               const float* __restrict__ resid,
                               float* __restrict__ C,
                               int M, int N, int K, int relu) {
  int wave = threadIdx.x >> 5;
  int lane = threadIdx.x & 31;
  int mt = blockIdx.x * 4 + wave;
  int nt = blockIdx.y;
  int half = lane >> 4, l16 = lane & 15;
  int row = mt * 16 + l16;
  int col = nt * 16 + l16;
  bool cok = col < N;
  int rowc = row < M ? row : M - 1;
  int colc = cok ? col : N - 1;
  const float* Ar = A + (long)rowc * K;

  v8f acc;
#pragma unroll
  for (int i = 0; i < 8; ++i) acc[i] = 0.f;

  int kmain = K & ~31;
  for (int k0 = 0; k0 < kmain; k0 += 32) {
    if (k0 + 32 < kmain) __builtin_prefetch(Ar + k0 + 32, 0, 1);
    const float4* p0 = (const float4*)(Ar + k0 + half * 8);
    const float4* p1 = (const float4*)(Ar + k0 + 16 + half * 8);
    float4 a0 = p0[0], a1 = p0[1], a2 = p1[0], a3 = p1[1];
    v16h a, b;
#pragma unroll
    for (int e = 0; e < 16; ++e)
      b[e] = (_Float16)W[(long)(k0 + frag_k(e, half)) * N + colc];
    a[0] = (_Float16)a0.x; a[1] = (_Float16)a0.y; a[2] = (_Float16)a0.z; a[3] = (_Float16)a0.w;
    a[4] = (_Float16)a1.x; a[5] = (_Float16)a1.y; a[6] = (_Float16)a1.z; a[7] = (_Float16)a1.w;
    a[8] = (_Float16)a2.x; a[9] = (_Float16)a2.y; a[10] = (_Float16)a2.z; a[11] = (_Float16)a2.w;
    a[12] = (_Float16)a3.x; a[13] = (_Float16)a3.y; a[14] = (_Float16)a3.z; a[15] = (_Float16)a3.w;
    acc = __builtin_amdgcn_wmma_f32_16x16x32_f16(false, a, false, b,
                                                 (short)0, acc, false, false);
  }
  if (kmain < K) {            // tail chunk (K=100 case)
    v16h a, b;
#pragma unroll
    for (int e = 0; e < 16; ++e) {
      int k = kmain + frag_k(e, half);
      int kc = k < K ? k : K - 1;
      float av = Ar[kc];
      a[e] = (_Float16)(k < K ? av : 0.f);
      b[e] = (_Float16)W[(long)kc * N + colc];
    }
    acc = __builtin_amdgcn_wmma_f32_16x16x32_f16(false, a, false, b,
                                                 (short)0, acc, false, false);
  }
  if (cok) {
    float bia = bias ? bias[col] : 0.f;
#pragma unroll
    for (int r = 0; r < 8; ++r) {
      int m = mt * 16 + half * 8 + r;
      if (m < M) {
        float v = acc[r] + bia;
        if (relu) v = fmaxf(v, 0.f);
        if (resid) v += resid[(long)m * N + col];
        C[(long)m * N + col] = v;
      }
    }
  }
}

// ---------------------------------------------------------------------------
// Attention core per (batch, head): 12x25 q,k,v -> softmax(qk^T/5) @ v
// grid = 1024, block = 32
// ---------------------------------------------------------------------------
__global__ void attn_core_kernel(const float* __restrict__ q,
                                 const float* __restrict__ k,
                                 const float* __restrict__ v,
                                 float* __restrict__ o) {
  int b = blockIdx.x >> 2;
  int h = blockIdx.x & 3;
  int lane = threadIdx.x;
  __shared__ float qa[12][25], ka[12][25], va[12][25], s[12][12];
  for (int i = lane; i < 300; i += 32) {
    int t = i / 25, d = i % 25;
    long base = ((long)(b * TT + t) * NN) + h * 25 + d;
    qa[t][d] = q[base]; ka[t][d] = k[base]; va[t][d] = v[base];
  }
  __syncthreads();
  const float scale = 0.2f;  // 1/sqrt(25)
  for (int i = lane; i < 144; i += 32) {
    int tq = i / 12, tk = i % 12;
    float acc = 0.f;
#pragma unroll
    for (int d = 0; d < 25; ++d) acc += qa[tq][d] * ka[tk][d];
    s[tq][tk] = acc * scale;
  }
  __syncthreads();
  if (lane < 12) {
    float mx = -1e30f;
#pragma unroll
    for (int j = 0; j < 12; ++j) mx = fmaxf(mx, s[lane][j]);
    float sum = 0.f;
#pragma unroll
    for (int j = 0; j < 12; ++j) { float e = __expf(s[lane][j] - mx); s[lane][j] = e; sum += e; }
    float inv = 1.f / sum;
#pragma unroll
    for (int j = 0; j < 12; ++j) s[lane][j] *= inv;
  }
  __syncthreads();
  for (int i = lane; i < 300; i += 32) {
    int t = i / 25, d = i % 25;
    float acc = 0.f;
#pragma unroll
    for (int tk = 0; tk < 12; ++tk) acc += s[t][tk] * va[tk][d];
    o[((long)(b * TT + t) * NN) + h * 25 + d] = acc;
  }
}

// ---------------------------------------------------------------------------
// out[b] = mean_t sigmoid(x2[b,t,:] . lin_w + lin_b). grid = 256, block = 32
// ---------------------------------------------------------------------------
__global__ void final_out_kernel(const float* __restrict__ x2,
                                 const float* __restrict__ lw,
                                 const float* __restrict__ lb,
                                 float* __restrict__ out) {
  int b = blockIdx.x;
  int lane = threadIdx.x;
  float accs = 0.f;
  for (int t = 0; t < TT; ++t) {
    float p = 0.f;
    for (int n = lane; n < NN; n += 32)
      p += x2[((long)(b * TT + t)) * NN + n] * lw[n];
    p = warp_sum(p);
    accs += 1.f / (1.f + __expf(-(p + lb[0])));
  }
  if (lane == 0) out[b] = accs / 12.f;
}

// ---------------------------------------------------------------------------
extern "C" void kernel_launch(void* const* d_in, const int* in_sizes, int n_in,
                              void* d_out, int out_size, void* d_ws, size_t ws_size,
                              hipStream_t stream) {
  const float* x    = (const float*)d_in[0];
  const float* adj  = (const float*)d_in[1];
  const float* c    = (const float*)d_in[2];
  const float* dyw[4] = {(const float*)d_in[3], (const float*)d_in[5],
                         (const float*)d_in[7], (const float*)d_in[9]};
  const float* dyb[4] = {(const float*)d_in[4], (const float*)d_in[6],
                         (const float*)d_in[8], (const float*)d_in[10]};
  const float* bng  = (const float*)d_in[11];
  const float* bnb  = (const float*)d_in[12];
  const float* n1a  = (const float*)d_in[13];
  const float* n1b  = (const float*)d_in[14];
  const float* n2a  = (const float*)d_in[15];
  const float* n2b  = (const float*)d_in[16];
  const float* wq   = (const float*)d_in[17];
  const float* bq   = (const float*)d_in[18];
  const float* wk   = (const float*)d_in[19];
  const float* bk   = (const float*)d_in[20];
  const float* wv   = (const float*)d_in[21];
  const float* bv   = (const float*)d_in[22];
  const float* wo   = (const float*)d_in[23];
  const float* bo   = (const float*)d_in[24];
  const float* fw1  = (const float*)d_in[25];
  const float* fb1  = (const float*)d_in[26];
  const float* fw2  = (const float*)d_in[27];
  const float* fb2  = (const float*)d_in[28];
  const float* norma = (const float*)d_in[29];
  const float* normb = (const float*)d_in[30];
  const float* linw = (const float*)d_in[31];
  const float* linb = (const float*)d_in[32];
  float* out = (float*)d_out;

  // Workspace layout:
  //   X   : f32 raw activations, ROWS x 256 max -> 314,572,800 B
  //   Y   : f16 HW buffer,       ROWS x 256     -> 157,286,400 B (float-reused later)
  //   mu/rstd : 1200 floats each
  char* ws = (char*)d_ws;
  float*    X  = (float*)ws;
  _Float16* Y  = (_Float16*)(ws + 314572800UL);
  float*    mu = (float*)(ws + 471859200UL);
  float*    rs = (float*)(ws + 471864000UL);
  float*    Yf = (float*)(ws + 314572800UL);   // attention-stage scratch (Y reuse)
  float* hseq = Yf;
  float* x2   = Yf + 1 * ROWS;
  float* qb   = Yf + 2 * ROWS;
  float* kb   = Yf + 3 * ROWS;
  float* vb   = Yf + 4 * ROWS;
  float* ob   = Yf + 5 * ROWS;
  float* t1   = Yf + 6 * ROWS;                 // 3072 x 2048

  pack_h0_kernel<<<1200, 256, 0, stream>>>(x, c, X);

  const int fin[4]  = {4, 64, 128, 256};
  const int fout[4] = {64, 128, 256, 1};
  for (int L = 0; L < 4; ++L) {
    int ntl = (fout[L] + 15) / 16;
    dim3 g1(4800, ntl);
    gemm_hw_kernel<<<g1, 128, 0, stream>>>(
        X, dyw[L], Y, mu, rs,
        L ? bng + (L - 1) * NN : bng,
        L ? bnb + (L - 1) * NN : bnb,
        ROWS, fout[L], fin[L], L ? 1 : 0);
    dim3 g2(BT, 7, (ntl + 3) / 4);
    gemm_adj_kernel<<<g2, 128, 0, stream>>>(adj, Y, dyb[L], X, fout[L]);
    bn_stats_kernel<<<1200, 256, 0, stream>>>(X, mu, rs, fout[L]);
  }
  finalize_dy_kernel<<<1200, 256, 0, stream>>>(X, mu, rs, bng + 300, bnb + 300, hseq);

  for (int j = 0; j < 2; ++j) {
    ln_rows_kernel<<<768, 128, 0, stream>>>(hseq, n1a + j * NN, n1b + j * NN, x2);
    gemm_rm_kernel<<<dim3(48, 7), 128, 0, stream>>>(x2, wq + j * 10000, bq + j * NN,
                                                    nullptr, qb, BT, NN, NN, 0);
    gemm_rm_kernel<<<dim3(48, 7), 128, 0, stream>>>(x2, wk + j * 10000, bk + j * NN,
                                                    nullptr, kb, BT, NN, NN, 0);
    gemm_rm_kernel<<<dim3(48, 7), 128, 0, stream>>>(x2, wv + j * 10000, bv + j * NN,
                                                    nullptr, vb, BT, NN, NN, 0);
    attn_core_kernel<<<1024, 32, 0, stream>>>(qb, kb, vb, ob);
    gemm_rm_kernel<<<dim3(48, 7), 128, 0, stream>>>(ob, wo + j * 10000, bo + j * NN,
                                                    hseq, hseq, BT, NN, NN, 0);
    ln_rows_kernel<<<768, 128, 0, stream>>>(hseq, n2a + j * NN, n2b + j * NN, x2);
    gemm_rm_kernel<<<dim3(48, 128), 128, 0, stream>>>(x2, fw1 + j * NN * 2048,
                                                      fb1 + j * 2048, nullptr, t1,
                                                      BT, 2048, NN, 1);
    gemm_rm_kernel<<<dim3(48, 7), 128, 0, stream>>>(t1, fw2 + j * 2048 * NN,
                                                    fb2 + j * NN, hseq, hseq,
                                                    BT, NN, 2048, 0);
  }
  ln_rows_kernel<<<768, 128, 0, stream>>>(hseq, norma, normb, x2);
  final_out_kernel<<<256, 32, 0, stream>>>(x2, linw, linb, out);
}
